// Multi_Head_Attention_46119358825152
// MI455X (gfx1250) — compile-verified, matrix operand order checked by reading, not run on hardware
//
#include <hip/hip_runtime.h>

typedef __bf16 bf16_t;
typedef __attribute__((ext_vector_type(16))) __bf16 v16bf;
typedef __attribute__((ext_vector_type(8)))  float  v8f;

#define B_  2
#define S_  2048
#define D_  1024
#define H_  16
#define DH_ 64

static __device__ __forceinline__ v8f zero_v8f() {
  v8f z = {0.f, 0.f, 0.f, 0.f, 0.f, 0.f, 0.f, 0.f};
  return z;
}

static __device__ __forceinline__ v8f wmma_bf16(v16bf a, v16bf b, v8f c) {
  return __builtin_amdgcn_wmma_f32_16x16x32_bf16(false, a, false, b, (short)0, c,
                                                 false, false);
}

// pack two f32 into one dword of two bf16 (lo = a, hi = b) -> v_cvt_pk_bf16_f32
static __device__ __forceinline__ unsigned pack_bf16(float a, float b) {
  bf16_t lo = (bf16_t)a, hi = (bf16_t)b;
  unsigned short ulo = __builtin_bit_cast(unsigned short, lo);
  unsigned short uhi = __builtin_bit_cast(unsigned short, hi);
  return (unsigned)ulo | ((unsigned)uhi << 16);
}

// Fragment from row-major bf16 source with contiguous K pairs.
// A-fragment (rows = M), or B-fragment when source rows are columns of B.
static __device__ __forceinline__ v16bf frag_rows_bf16(const bf16_t* base, int ld,
                                                       int row0, int k0, int lane) {
  const int m = lane & 15, half = (lane >> 4) & 1;
  union { v16bf v; unsigned u[8]; } f;
  const unsigned* p = (const unsigned*)(base + (size_t)(row0 + m) * ld + k0);
#pragma unroll
  for (int r = 0; r < 8; ++r) {
    const int kk = ((r >> 2) << 4) + (half << 3) + ((r & 3) << 1);
    f.u[r] = p[kk >> 1];  // contiguous dwords -> ds/global b128 clauses
  }
  return f.v;
}

// A-fragment from f32 source (contiguous 8-float runs -> global b128 loads).
static __device__ __forceinline__ v16bf frag_rows_f32(const float* base, int ld,
                                                      int row0, int k0, int lane) {
  const int m = lane & 15, half = (lane >> 4) & 1;
  v16bf v;
  const float* p = base + (size_t)(row0 + m) * ld + k0;
#pragma unroll
  for (int r = 0; r < 8; ++r) {
    const int kk = ((r >> 2) << 4) + (half << 3) + ((r & 3) << 1);
    v[2 * r]     = (bf16_t)p[kk];
    v[2 * r + 1] = (bf16_t)p[kk + 1];
  }
  return v;
}

// Stage W[k0..k0+32) x [n0..n0+128) transposed+packed into LDS:
// dword (n*16 + kp) = {bf16 W[k0+2kp][n], bf16 W[k0+2kp+1][n]}.
// Each thread: 2 contiguous 16-float row segments (b128 clauses),
// 16 cvt_pk, 16 ds_store_b32.
static __device__ __forceinline__ void stage_wt(unsigned* sBtU, const float* W,
                                                int k0, int n0, int tid) {
  const int ng = tid & 7, kp = tid >> 3;     // 8 n-groups of 16, 16 k-pairs
  const float* r0 = W + (size_t)(k0 + 2 * kp) * D_ + n0 + ng * 16;
  const float* r1 = r0 + D_;
#pragma unroll
  for (int j = 0; j < 16; ++j) {
    sBtU[(ng * 16 + j) * 16 + kp] = pack_bf16(r0[j], r1[j]);
  }
}

// ---------------------------------------------------------------------------
// Kernel 1: Y = X @ W + b (f32 in, bf16 WMMA, bf16 out).
// 64x128 tile, 128 threads (4 waves).  A-fragments straight from global;
// W tile staged transposed (k-pair packed) in LDS.
// transposed==0: out[b,h,s,dh]   (Q, K)
// transposed==1: out[b,h,dh,s]   (V^T for the P@V GEMM)
// ---------------------------------------------------------------------------
__global__ __launch_bounds__(128) void proj_qkv_kernel(
    const float* __restrict__ X, const float* __restrict__ W,
    const float* __restrict__ bias, bf16_t* __restrict__ out, int transposed) {
  __shared__ unsigned sBtU[128 * 16];            // [n][kpair] -> 8 KB
  const bf16_t* sBt = (const bf16_t*)sBtU;       // [n][k], ld = 32
  const int tid  = threadIdx.x;
  const int lane = tid & 31, w = tid >> 5;
  const int row0 = blockIdx.x * 64;              // row in [B*S)
  const int n0   = blockIdx.y * 128;             // output feature
  v8f acc[8];
#pragma unroll
  for (int nt = 0; nt < 8; ++nt) acc[nt] = zero_v8f();

  for (int kt = 0; kt < D_ / 32; ++kt) {
    const int k0 = kt * 32;
    stage_wt(sBtU, W, k0, n0, tid);
    if (kt + 1 < D_ / 32) {
      __builtin_prefetch(&W[(size_t)(k0 + 32 + (tid >> 3)) * D_ + n0 + (tid & 7) * 16], 0, 1);
      __builtin_prefetch(&X[(size_t)(row0 + (tid >> 1)) * D_ + k0 + 32 + (tid & 1) * 16], 0, 1);
    }
    __syncthreads();
    v16bf aF = frag_rows_f32(X, D_, row0 + w * 16, k0, lane);
#pragma unroll
    for (int nt = 0; nt < 8; ++nt) {
      v16bf bF = frag_rows_bf16(sBt, 32, nt * 16, 0, lane);
      acc[nt] = wmma_bf16(aF, bF, acc[nt]);
    }
    __syncthreads();
  }

  const int n = lane & 15, half = (lane >> 4) & 1;
#pragma unroll
  for (int nt = 0; nt < 8; ++nt) {
#pragma unroll
    for (int r = 0; r < 8; ++r) {
      int m  = half * 8 + r;
      int gr = row0 + w * 16 + m;          // b*S + s
      int gc = n0 + nt * 16 + n;           // feature d
      float v = acc[nt][r] + bias[gc];
      int b = gr >> 11, s = gr & (S_ - 1);
      int h = gc >> 6,  dh = gc & (DH_ - 1);
      size_t idx = transposed
          ? (((size_t)(b * H_ + h) * DH_) + dh) * S_ + s
          : (((size_t)(b * H_ + h) * S_) + s) * DH_ + dh;
      out[idx] = (bf16_t)v;
    }
  }
}

// ---------------------------------------------------------------------------
// Kernel 2: attention for one (b, h, 16-query-row tile).  128 threads, 4 waves.
// ---------------------------------------------------------------------------
__global__ __launch_bounds__(128) void attn_kernel(
    const bf16_t* __restrict__ Q, const bf16_t* __restrict__ K,
    const bf16_t* __restrict__ Vt,     // [B,H,DH,S]
    float* __restrict__ attn, bf16_t* __restrict__ ctx) {
  __shared__ float sRed[16][8];
  const int tid  = threadIdx.x;
  const int lane = tid & 31, w = tid >> 5;
  const int q0 = blockIdx.x * 16;
  const int h  = blockIdx.y;
  const int b  = blockIdx.z;
  const size_t headOff = ((size_t)(b * H_ + h)) * S_ * DH_;
  const bf16_t* Qh  = Q + headOff;
  const bf16_t* Kh  = K + headOff;
  const bf16_t* Vth = Vt + headOff;    // rows = dh (len S)
  float* aRow = attn + (((size_t)(b * H_ + h)) * S_ + q0) * S_;

  v16bf aQ0 = frag_rows_bf16(Qh, DH_, q0, 0, lane);
  v16bf aQ1 = frag_rows_bf16(Qh, DH_, q0, 32, lane);

  const int n = lane & 15, half = (lane >> 4) & 1;

  // ---- phase 1: scaled+masked logits -> attn slab ----
  for (int nt = w; nt < S_ / 16; nt += 4) {
    v16bf bK0 = frag_rows_bf16(Kh, DH_, nt * 16, 0, lane);   // B = K^T
    v16bf bK1 = frag_rows_bf16(Kh, DH_, nt * 16, 32, lane);
    v8f acc = zero_v8f();
    acc = wmma_bf16(aQ0, bK0, acc);
    acc = wmma_bf16(aQ1, bK1, acc);
#pragma unroll
    for (int r = 0; r < 8; ++r) {
      int m = half * 8 + r;
      int j = nt * 16 + n;
      float v = acc[r] * 0.125f;           // 1/sqrt(DH)
      if (j > q0 + m) v -= 1e9f;           // causal mask
      aRow[(size_t)m * S_ + j] = v;
    }
  }
  __threadfence_block();
  __syncthreads();

  // ---- phase 2: row max / sum (16 rows x 8 segments of 256, float4) ----
  const int row = tid >> 3, seg = tid & 7;
  const float4* rp4 = (const float4*)(aRow + (size_t)row * S_ + seg * 256);
  float mx = -3.4e38f;
  for (int i = 0; i < 64; ++i) {
    float4 t = rp4[i];
    mx = fmaxf(mx, fmaxf(fmaxf(t.x, t.y), fmaxf(t.z, t.w)));
  }
  sRed[row][seg] = mx;
  __syncthreads();
  float rmx = sRed[row][0];
#pragma unroll
  for (int i = 1; i < 8; ++i) rmx = fmaxf(rmx, sRed[row][i]);
  __syncthreads();
  float sm = 0.f;
  for (int i = 0; i < 64; ++i) {
    float4 t = rp4[i];
    sm += __expf(t.x - rmx) + __expf(t.y - rmx) +
          __expf(t.z - rmx) + __expf(t.w - rmx);
  }
  sRed[row][seg] = sm;
  __syncthreads();
  float rsum = 0.f;
#pragma unroll
  for (int i = 0; i < 8; ++i) rsum += sRed[row][i];
  const float rinv = 1.0f / rsum;

  // ---- phase 3: normalize in place ----
  float4* wp4 = (float4*)(aRow + (size_t)row * S_ + seg * 256);
  for (int i = 0; i < 64; ++i) {
    float4 t = wp4[i];
    t.x = __expf(t.x - rmx) * rinv;
    t.y = __expf(t.y - rmx) * rinv;
    t.z = __expf(t.z - rmx) * rinv;
    t.w = __expf(t.w - rmx) * rinv;
    wp4[i] = t;
  }
  __threadfence_block();
  __syncthreads();

  // ---- phase 4: ctx = P @ V, wave w owns 16 output columns ----
  v8f acc = zero_v8f();
  for (int kc = 0; kc < S_ / 32; ++kc) {
    v16bf aP = frag_rows_f32(aRow, S_, 0, kc * 32, lane);
    v16bf bV = frag_rows_bf16(Vth, S_, w * 16, kc * 32, lane);  // rows of V^T
    acc = wmma_bf16(aP, bV, acc);
  }
#pragma unroll
  for (int r = 0; r < 8; ++r) {
    int m = half * 8 + r;
    size_t idx = ((size_t)(b * S_) + q0 + m) * D_ + h * DH_ + w * 16 + n;
    ctx[idx] = (bf16_t)acc[r];
  }
}

// ---------------------------------------------------------------------------
// Kernel 3: out = ctx(bf16) @ Wo + bo, f32 output into d_out.
// ---------------------------------------------------------------------------
__global__ __launch_bounds__(128) void out_proj_kernel(
    const bf16_t* __restrict__ Xc, const float* __restrict__ W,
    const float* __restrict__ bias, float* __restrict__ out) {
  __shared__ unsigned sBtU[128 * 16];
  const bf16_t* sBt = (const bf16_t*)sBtU;
  const int tid  = threadIdx.x;
  const int lane = tid & 31, w = tid >> 5;
  const int row0 = blockIdx.x * 64;
  const int n0   = blockIdx.y * 128;
  v8f acc[8];
#pragma unroll
  for (int nt = 0; nt < 8; ++nt) acc[nt] = zero_v8f();

  for (int kt = 0; kt < D_ / 32; ++kt) {
    const int k0 = kt * 32;
    stage_wt(sBtU, W, k0, n0, tid);
    if (kt + 1 < D_ / 32) {
      __builtin_prefetch(&W[(size_t)(k0 + 32 + (tid >> 3)) * D_ + n0 + (tid & 7) * 16], 0, 1);
      __builtin_prefetch(&Xc[(size_t)(row0 + (tid >> 1)) * D_ + k0 + 32 + (tid & 1) * 16], 0, 1);
    }
    __syncthreads();
    v16bf aF = frag_rows_bf16(Xc, D_, row0 + w * 16, k0, lane);
#pragma unroll
    for (int nt = 0; nt < 8; ++nt) {
      v16bf bF = frag_rows_bf16(sBt, 32, nt * 16, 0, lane);
      acc[nt] = wmma_bf16(aF, bF, acc[nt]);
    }
    __syncthreads();
  }

  const int n = lane & 15, half = (lane >> 4) & 1;
#pragma unroll
  for (int nt = 0; nt < 8; ++nt) {
#pragma unroll
    for (int r = 0; r < 8; ++r) {
      int m  = half * 8 + r;
      int gr = row0 + w * 16 + m;
      int gc = n0 + nt * 16 + n;
      out[(size_t)gr * D_ + gc] = acc[nt][r] + bias[gc];
    }
  }
}

extern "C" void kernel_launch(void* const* d_in, const int* in_sizes, int n_in,
                              void* d_out, int out_size, void* d_ws, size_t ws_size,
                              hipStream_t stream) {
  (void)in_sizes; (void)n_in; (void)out_size; (void)ws_size;
  const float* v  = (const float*)d_in[0];
  const float* k  = (const float*)d_in[1];
  const float* q  = (const float*)d_in[2];
  // d_in[3] = mask (causal, applied analytically)
  const float* Wq = (const float*)d_in[4];
  const float* bq = (const float*)d_in[5];
  const float* Wk = (const float*)d_in[6];
  const float* bk = (const float*)d_in[7];
  const float* Wv = (const float*)d_in[8];
  const float* bv = (const float*)d_in[9];
  const float* Wo = (const float*)d_in[10];
  const float* bo = (const float*)d_in[11];

  float* out  = (float*)d_out;
  float* attn = out + (size_t)B_ * S_ * D_;

  const size_t perMat = (size_t)B_ * H_ * S_ * DH_;   // 4M bf16 elements
  bf16_t* Qbf  = (bf16_t*)d_ws;
  bf16_t* Kbf  = Qbf + perMat;
  bf16_t* Vtbf = Kbf + perMat;                         // [B,H,DH,S]
  bf16_t* ctx  = Vtbf + perMat;                        // [B,S,D] bf16

  dim3 gProj(B_ * S_ / 64, D_ / 128);
  proj_qkv_kernel<<<gProj, 128, 0, stream>>>(q, Wq, bq, Qbf, 0);
  proj_qkv_kernel<<<gProj, 128, 0, stream>>>(k, Wk, bk, Kbf, 0);
  proj_qkv_kernel<<<gProj, 128, 0, stream>>>(v, Wv, bv, Vtbf, 1);
  attn_kernel<<<dim3(S_ / 16, H_, B_), 128, 0, stream>>>(Qbf, Kbf, Vtbf, attn, ctx);
  out_proj_kernel<<<gProj, 128, 0, stream>>>(ctx, Wo, bo, out);
}
